// STTransformerBlock_5351529250841
// MI455X (gfx1250) — matrix operand order repaired and checked by reading, compiled, not run on hardware
//
#include <hip/hip_runtime.h>
#include <hip/hip_bf16.h>

// ---------------------------------------------------------------------------
// Problem constants (reference: B=8, T=12, N=325, D=256, H=8, HD=32, FF=1024)
// ---------------------------------------------------------------------------
#define BB   8
#define TT   12
#define NN   325
#define DD   256
#define HH   8
#define HDIM 32
#define FFD  1024
#define MROWS (BB * TT * NN)           // 31200
#define ATT_SCALE 0.17677669529663687f // 32^-0.5
#define NEG_BIG  (-1.0e30f)

// ---------------------------------------------------------------------------
// WMMA types / helpers (gfx1250 wave32, v_wmma_f32_16x16x32_bf16)
// ---------------------------------------------------------------------------
typedef __attribute__((ext_vector_type(16))) __bf16 v16bf;
typedef __attribute__((ext_vector_type(2)))  __bf16 v2bf;
typedef __attribute__((ext_vector_type(2)))  float  v2f;
typedef __attribute__((ext_vector_type(8)))  float  v8f;

union AFrag { unsigned int u[8]; v16bf v; };
union CFrag { float f[8]; v8f v; };
union H2    { unsigned short s[2]; unsigned int u; }; // d16 lo/hi assembly

__device__ __forceinline__ v8f wmma_bf16(v16bf a, v16bf b, v8f c) {
  return __builtin_amdgcn_wmma_f32_16x16x32_bf16(
      /*neg_a=*/false, a, /*neg_b=*/false, b,
      /*c_mod=*/(short)0, c, /*reuse_a=*/false, /*reuse_b=*/false);
}

// float pair -> packed bf16 (lowers to v_cvt_pk_bf16_f32 on gfx1250)
__device__ __forceinline__ unsigned int packbf2(float lo, float hi) {
  v2f f; f.x = lo; f.y = hi;
  union { v2bf v; unsigned int u; } cv;
  cv.v = __builtin_convertvector(f, v2bf);
  return cv.u;
}

// A-matrix (16x32, 16-bit) K index for VGPR pair p, lane-half `half`
// (ISA 7.12.2: V0..3 -> K {0..7 | 8..15}, V4..7 -> K {16..23 | 24..31})
__device__ __forceinline__ int ka_idx(int p, int half) {
  return ((p & 4) << 2) + (half << 3) + ((p & 3) << 1);
}
// B-matrix (32x16, 16-bit) K index: same order as A (V0..3 -> K {0..7 | 8..15}, V4..7 -> K {16..23 | 24..31})
__device__ __forceinline__ int kb_idx(int p, int half) {
  return ((p & 4) << 2) + (half << 3) + ((p & 3) << 1);
}

// ---------------------------------------------------------------------------
// Generic GEMM:  C[M,N] = act(A[M,K] @ W[K,N] + bias[N]),  act: 0=id, 1=GELU
// Block: 256 thr = 8 waves; block tile 128(M) x 64(N); wave tile 32x32:
// 2 A-frags x 2 B-frags -> 4 WMMAs per K-step per wave.
// Staging issues all 6 global_load_b128 before any convert/store so the
// loads pipeline (loadcnt counts down instead of draining per load).
// ---------------------------------------------------------------------------
#define LDT 40   // LDS row stride (bf16 elems): 80B -> bank-spread, 16B-align ok

__global__ __launch_bounds__(256) void gemm_bf16_kernel(
    const float* __restrict__ A, const float* __restrict__ W,
    const float* __restrict__ bias, float* __restrict__ C,
    int M, int K, int N, int act)
{
  __shared__ __align__(16) __bf16 sA[128 * LDT]; // [m][k]
  __shared__ __align__(16) __bf16 sW[64 * LDT];  // [n][k] (transposed)

  const int tid  = threadIdx.x;
  const int wave = tid >> 5, lane = tid & 31;
  const int half = lane >> 4, l16 = lane & 15;
  const int waveM = wave >> 1;       // 0..3 -> 32-row band
  const int waveN = wave & 1;        // 0..1 -> 32-col band
  const int blockM = blockIdx.y * 128;
  const int blockN = blockIdx.x * 64;

  // per-thread load coordinates
  const int amRow = tid >> 3;               // 0..31 (+32 per iter)
  const int amCol = (tid & 7) << 2;         // 0,4,..,28
  const int wkRow = tid >> 4;               // 0..15 (+16 on 2nd iter)
  const int wnCol = (tid & 15) << 2;        // 0,4,..,60

  CFrag a00, a01, a10, a11;
#pragma unroll
  for (int i = 0; i < 8; i++) { a00.f[i] = 0.f; a01.f[i] = 0.f; a10.f[i] = 0.f; a11.f[i] = 0.f; }

  for (int k0 = 0; k0 < K; k0 += 32) {
    // ---- issue ALL tile loads first (6x global_load_b128 in flight) ----
    float4 fa[4];
    bool   va[4];
#pragma unroll
    for (int i = 0; i < 4; i++) {
      int gm = blockM + amRow + i * 32;
      va[i] = (gm < M);
      int cm = va[i] ? gm : (M - 1);
      fa[i] = *(const float4*)&A[(size_t)cm * K + k0 + amCol];
    }
    float4 fw[2];
#pragma unroll
    for (int i = 0; i < 2; i++) {
      int kk = wkRow + i * 16;
      fw[i] = *(const float4*)&W[(size_t)(k0 + kk) * N + blockN + wnCol];
    }
    // prefetch next K tile (global_prefetch_b8)
    if (k0 + 32 < K) {
      int gm = blockM + amRow; if (gm >= M) gm = M - 1;
      __builtin_prefetch(&A[(size_t)gm * K + k0 + 32 + amCol], 0, 1);
      __builtin_prefetch(&W[(size_t)(k0 + 32 + wkRow) * N + blockN + wnCol], 0, 1);
    }
    // ---- convert + store to LDS ----
#pragma unroll
    for (int i = 0; i < 4; i++) {
      float4 f = fa[i];
      if (!va[i]) { f.x = 0.f; f.y = 0.f; f.z = 0.f; f.w = 0.f; }
      uint2 pk; pk.x = packbf2(f.x, f.y); pk.y = packbf2(f.z, f.w);
      *(uint2*)&sA[(amRow + i * 32) * LDT + amCol] = pk;
    }
#pragma unroll
    for (int i = 0; i < 2; i++) {
      int kk = wkRow + i * 16;
      sW[(wnCol + 0) * LDT + kk] = (__bf16)fw[i].x;
      sW[(wnCol + 1) * LDT + kk] = (__bf16)fw[i].y;
      sW[(wnCol + 2) * LDT + kk] = (__bf16)fw[i].z;
      sW[(wnCol + 3) * LDT + kk] = (__bf16)fw[i].w;
    }
    __syncthreads();

    AFrag af0, af1, bf0, bf1;
    const int arow0 = waveM * 32 + l16;
    const int brow0 = waveN * 32 + l16;
#pragma unroll
    for (int p = 0; p < 8; p++) {
      int kaof = ka_idx(p, half);
      int kbof = kb_idx(p, half);
      af0.u[p] = *(const unsigned int*)&sA[arow0 * LDT + kaof];
      af1.u[p] = *(const unsigned int*)&sA[(arow0 + 16) * LDT + kaof];
      bf0.u[p] = *(const unsigned int*)&sW[brow0 * LDT + kbof];
      bf1.u[p] = *(const unsigned int*)&sW[(brow0 + 16) * LDT + kbof];
    }
    a00.v = wmma_bf16(af0.v, bf0.v, a00.v);
    a01.v = wmma_bf16(af0.v, bf1.v, a01.v);
    a10.v = wmma_bf16(af1.v, bf0.v, a10.v);
    a11.v = wmma_bf16(af1.v, bf1.v, a11.v);
    __syncthreads();
  }

  const int gn0 = blockN + waveN * 32 + l16;
  const int gn1 = gn0 + 16;
  const float bv0 = bias ? bias[gn0] : 0.0f;
  const float bv1 = bias ? bias[gn1] : 0.0f;
#pragma unroll
  for (int r = 0; r < 8; r++) {
    int gm0 = blockM + waveM * 32 + r + half * 8;
    int gm1 = gm0 + 16;
    float v00 = a00.f[r] + bv0, v01 = a01.f[r] + bv1;
    float v10 = a10.f[r] + bv0, v11 = a11.f[r] + bv1;
    if (act == 1) {
      v00 = 0.5f * v00 * (1.0f + erff(v00 * 0.70710678118654752f));
      v01 = 0.5f * v01 * (1.0f + erff(v01 * 0.70710678118654752f));
      v10 = 0.5f * v10 * (1.0f + erff(v10 * 0.70710678118654752f));
      v11 = 0.5f * v11 * (1.0f + erff(v11 * 0.70710678118654752f));
    }
    if (gm0 < M) { C[(size_t)gm0 * N + gn0] = v00; C[(size_t)gm0 * N + gn1] = v01; }
    if (gm1 < M) { C[(size_t)gm1 * N + gn0] = v10; C[(size_t)gm1 * N + gn1] = v11; }
  }
}

// ---------------------------------------------------------------------------
// Spatial flash attention: groups g = b*T+t (96), heads 8, L = 325, adj mask.
// Block = 8 waves; wave owns 16 query rows; block covers 128 rows.
// grid = (3, H=8, G=96)
// ---------------------------------------------------------------------------
#define SKV 40

__global__ __launch_bounds__(256) void spat_attn_kernel(
    const float* __restrict__ Q, const float* __restrict__ Kp,
    const float* __restrict__ Vp, const int* __restrict__ adj,
    float* __restrict__ O)
{
  __shared__ __align__(16) __bf16 sK[32 * SKV]; // [key][dim]
  __shared__ __align__(16) __bf16 sV[32 * SKV];
  __shared__ __align__(16) __bf16 sP[8][16 * 34]; // per-wave P staging

  const int tid = threadIdx.x;
  const int wave = tid >> 5, lane = tid & 31;
  const int half = lane >> 4, l16 = lane & 15;
  const int h = blockIdx.y, g = blockIdx.z;
  const int qbase = blockIdx.x * 128 + wave * 16;
  const int L = NN;
  const size_t gbase = (size_t)g * L * DD + (size_t)h * HDIM;

  // Q fragment (A-layout, 16 rows x HD=32)
  AFrag qf;
  {
    int qr = qbase + l16; if (qr >= L) qr = L - 1;
    const float* qp = Q + gbase + (size_t)qr * DD;
#pragma unroll
    for (int p = 0; p < 8; p++) {
      int kk = ka_idx(p, half);
      float2 qv = *(const float2*)&qp[kk];
      qf.u[p] = packbf2(qv.x, qv.y);
    }
  }

  CFrag o0, o1;
  float mrow[8], srow[8];
#pragma unroll
  for (int r = 0; r < 8; r++) { o0.f[r] = 0.f; o1.f[r] = 0.f; mrow[r] = NEG_BIG; srow[r] = 0.f; }

  // staging coords (one float4 per thread per array)
  const int skKey = tid >> 3;          // 0..31
  const int skDim = (tid & 7) << 2;    // 0,4,..,28

  const int nch = (L + 31) / 32; // 11
  for (int j = 0; j < nch; j++) {
    {
      int gk = j * 32 + skKey;
      int ck = (gk < L) ? gk : (L - 1);
      // issue both loads, then convert + store
      float4 kf = *(const float4*)&Kp[gbase + (size_t)ck * DD + skDim];
      float4 vf = *(const float4*)&Vp[gbase + (size_t)ck * DD + skDim];
      if (gk >= L) { kf.x=kf.y=kf.z=kf.w=0.f; vf.x=vf.y=vf.z=vf.w=0.f; }
      uint2 kp; kp.x = packbf2(kf.x, kf.y); kp.y = packbf2(kf.z, kf.w);
      uint2 vp; vp.x = packbf2(vf.x, vf.y); vp.y = packbf2(vf.z, vf.w);
      *(uint2*)&sK[skKey * SKV + skDim] = kp;
      *(uint2*)&sV[skKey * SKV + skDim] = vp;
    }
    __syncthreads();

    // Fragments: K^T (B for S = Q@K^T), V (B for P@V)
    AFrag kb0, kb1, vb0, vb1;
#pragma unroll
    for (int p = 0; p < 8; p++) {
      int kb = kb_idx(p, half);
      kb0.u[p] = *(const unsigned int*)&sK[l16 * SKV + kb];
      kb1.u[p] = *(const unsigned int*)&sK[(16 + l16) * SKV + kb];
      H2 t0, t1;
      t0.s[0] = *(const unsigned short*)&sV[kb * SKV + l16];
      t0.s[1] = *(const unsigned short*)&sV[(kb + 1) * SKV + l16];
      t1.s[0] = *(const unsigned short*)&sV[kb * SKV + 16 + l16];
      t1.s[1] = *(const unsigned short*)&sV[(kb + 1) * SKV + 16 + l16];
      vb0.u[p] = t0.u;
      vb1.u[p] = t1.u;
    }

    CFrag zc;
#pragma unroll
    for (int i = 0; i < 8; i++) zc.f[i] = 0.f;
    CFrag s0, s1;
    s0.v = wmma_bf16(qf.v, kb0.v, zc.v);
    s1.v = wmma_bf16(qf.v, kb1.v, zc.v);

    // scale + adjacency mask (branch-free clamp loads) + online softmax
    float p0[8], p1[8];
    const int c0 = j * 32 + l16, c1 = c0 + 16;
    const int cc0 = (c0 < L) ? c0 : (L - 1);
    const int cc1 = (c1 < L) ? c1 : (L - 1);
#pragma unroll
    for (int r = 0; r < 8; r++) {
      int qr = qbase + r + half * 8;
      int aq = (qr < L) ? qr : (L - 1);
      int m0 = adj[aq * NN + cc0];
      int m1 = adj[aq * NN + cc1];
      float v0 = s0.f[r] * ATT_SCALE;
      float v1 = s1.f[r] * ATT_SCALE;
      bool ok0 = (qr < L) && (c0 < L) && (m0 != 0);
      bool ok1 = (qr < L) && (c1 < L) && (m1 != 0);
      v0 = ok0 ? v0 : NEG_BIG;
      v1 = ok1 ? v1 : NEG_BIG;
      float mx = fmaxf(v0, v1);
      for (int d = 1; d < 16; d <<= 1) mx = fmaxf(mx, __shfl_xor(mx, d, 32));
      float mnew = fmaxf(mrow[r], mx);
      float alpha = __expf(mrow[r] - mnew);
      float e0 = __expf(v0 - mnew), e1 = __expf(v1 - mnew);
      float sm = e0 + e1;
      for (int d = 1; d < 16; d <<= 1) sm += __shfl_xor(sm, d, 32);
      srow[r] = srow[r] * alpha + sm;
      mrow[r] = mnew;
      o0.f[r] *= alpha; o1.f[r] *= alpha;
      p0[r] = e0; p1[r] = e1;
    }

    // stage P (C-layout -> LDS row-major -> A-layout fragment)
    __bf16* pw = &sP[wave][0];
#pragma unroll
    for (int r = 0; r < 8; r++) {
      int m = r + half * 8;
      pw[m * 34 + l16]      = (__bf16)p0[r];
      pw[m * 34 + 16 + l16] = (__bf16)p1[r];
    }
    AFrag pf;
#pragma unroll
    for (int p = 0; p < 8; p++)
      pf.u[p] = *(const unsigned int*)&pw[l16 * 34 + ka_idx(p, half)];

    o0.v = wmma_bf16(pf.v, vb0.v, o0.v);
    o1.v = wmma_bf16(pf.v, vb1.v, o1.v);
    __syncthreads();
  }

#pragma unroll
  for (int r = 0; r < 8; r++) {
    int qr = qbase + r + half * 8;
    if (qr < L) {
      float inv = (srow[r] > 0.f) ? 1.0f / srow[r] : 0.f;
      float* op = O + gbase + (size_t)qr * DD;
      op[l16]      = o0.f[r] * inv;
      op[16 + l16] = o1.f[r] * inv;
    }
  }
}

// ---------------------------------------------------------------------------
// Temporal causal attention: one block per (b,n) (2600), wave = head,
// T=12 queries/keys padded to 16/32. Single key chunk.
// ---------------------------------------------------------------------------
#define TKV 264

__global__ __launch_bounds__(256) void temp_attn_kernel(
    const float* __restrict__ Q, const float* __restrict__ Kp,
    const float* __restrict__ Vp, float* __restrict__ O)
{
  __shared__ __align__(16) __bf16 sK[32 * TKV]; // [t][c 0..255]
  __shared__ __align__(16) __bf16 sV[32 * TKV];
  __shared__ __align__(16) __bf16 sP[8][16 * 34];

  const int bn = blockIdx.x;
  const int b = bn / NN, n = bn % NN;
  const int tid = threadIdx.x;
  const int wave = tid >> 5, lane = tid & 31;
  const int half = lane >> 4, l16 = lane & 15;
  const int h = wave;

  // stage all heads' K/V rows (12 real + zero pad to 32); float4 loads
#pragma unroll
  for (int i = 0; i < 8; i++) {
    int idx = tid + i * 256;           // 0..2047 float4s
    int t = idx >> 6;                  // 0..31
    int c = (idx & 63) << 2;           // 0,4,..,252
    int ct = (t < TT) ? t : 0;
    size_t gr = ((size_t)(b * TT + ct) * NN + n) * DD + c;
    float4 kf = *(const float4*)&Kp[gr];
    float4 vf = *(const float4*)&Vp[gr];
    if (t >= TT) { kf.x=kf.y=kf.z=kf.w=0.f; vf.x=vf.y=vf.z=vf.w=0.f; }
    uint2 kp; kp.x = packbf2(kf.x, kf.y); kp.y = packbf2(kf.z, kf.w);
    uint2 vp; vp.x = packbf2(vf.x, vf.y); vp.y = packbf2(vf.z, vf.w);
    *(uint2*)&sK[t * TKV + c] = kp;
    *(uint2*)&sV[t * TKV + c] = vp;
  }
  __syncthreads();

  AFrag qf;
  {
    int qr = (l16 < TT) ? l16 : (TT - 1);
    const float* qp = Q + ((size_t)(b * TT + qr) * NN + n) * DD + h * HDIM;
#pragma unroll
    for (int p = 0; p < 8; p++) {
      int kk = ka_idx(p, half);
      float2 qv = *(const float2*)&qp[kk];
      qf.u[p] = packbf2(qv.x, qv.y);
    }
  }

  AFrag kb0, kb1, vb0, vb1;
#pragma unroll
  for (int p = 0; p < 8; p++) {
    int kb = kb_idx(p, half);
    kb0.u[p] = *(const unsigned int*)&sK[l16 * TKV + h * HDIM + kb];
    kb1.u[p] = *(const unsigned int*)&sK[(16 + l16) * TKV + h * HDIM + kb];
    H2 t0, t1;
    t0.s[0] = *(const unsigned short*)&sV[kb * TKV + h * HDIM + l16];
    t0.s[1] = *(const unsigned short*)&sV[(kb + 1) * TKV + h * HDIM + l16];
    t1.s[0] = *(const unsigned short*)&sV[kb * TKV + h * HDIM + 16 + l16];
    t1.s[1] = *(const unsigned short*)&sV[(kb + 1) * TKV + h * HDIM + 16 + l16];
    vb0.u[p] = t0.u;
    vb1.u[p] = t1.u;
  }

  CFrag zc;
#pragma unroll
  for (int i = 0; i < 8; i++) zc.f[i] = 0.f;
  CFrag s0, s1;
  s0.v = wmma_bf16(qf.v, kb0.v, zc.v);
  s1.v = wmma_bf16(qf.v, kb1.v, zc.v);

  float p0[8], p1[8], srow[8];
#pragma unroll
  for (int r = 0; r < 8; r++) {
    int qr = r + half * 8;
    int c0 = l16, c1 = 16 + l16;
    float v0 = s0.f[r] * ATT_SCALE;
    float v1 = s1.f[r] * ATT_SCALE;
    v0 = ((qr < TT) && (c0 < TT) && (c0 <= qr)) ? v0 : NEG_BIG;
    v1 = ((qr < TT) && (c1 < TT) && (c1 <= qr)) ? v1 : NEG_BIG;
    float mx = fmaxf(v0, v1);
    for (int d = 1; d < 16; d <<= 1) mx = fmaxf(mx, __shfl_xor(mx, d, 32));
    float e0 = __expf(v0 - mx), e1 = __expf(v1 - mx);
    float sm = e0 + e1;
    for (int d = 1; d < 16; d <<= 1) sm += __shfl_xor(sm, d, 32);
    srow[r] = sm;
    p0[r] = e0; p1[r] = e1;
  }

  __bf16* pw = &sP[wave][0];
#pragma unroll
  for (int r = 0; r < 8; r++) {
    int m = r + half * 8;
    pw[m * 34 + l16]      = (__bf16)p0[r];
    pw[m * 34 + 16 + l16] = (__bf16)p1[r];
  }
  AFrag pf;
#pragma unroll
  for (int p = 0; p < 8; p++)
    pf.u[p] = *(const unsigned int*)&pw[l16 * 34 + ka_idx(p, half)];

  CFrag o0, o1;
  o0.v = wmma_bf16(pf.v, vb0.v, zc.v);
  o1.v = wmma_bf16(pf.v, vb1.v, zc.v);

#pragma unroll
  for (int r = 0; r < 8; r++) {
    int qr = r + half * 8;
    if (qr < TT) {
      float inv = (srow[r] > 0.f) ? 1.0f / srow[r] : 0.f;
      float* op = O + ((size_t)(b * TT + qr) * NN + n) * DD + h * HDIM;
      op[l16]      = o0.f[r] * inv;
      op[16 + l16] = o1.f[r] * inv;
    }
  }
}

// ---------------------------------------------------------------------------
// y = LayerNorm(x + r) * w + b   (row width = 256, one wave per row, float4)
// Safe for Y == X (row fully read before written, one wave owns the row).
// ---------------------------------------------------------------------------
__global__ __launch_bounds__(256) void add_ln_kernel(
    const float* __restrict__ X, const float* __restrict__ R,
    const float* __restrict__ w, const float* __restrict__ b,
    float* __restrict__ Y, int M)
{
  const int row = blockIdx.x * 8 + (threadIdx.x >> 5);
  const int lane = threadIdx.x & 31;
  if (row >= M) return;
  const float4* xp = (const float4*)(X + (size_t)row * DD);
  const float4* rp = (const float4*)(R + (size_t)row * DD);
  float4 vals[2];
  float sum = 0.f;
#pragma unroll
  for (int i = 0; i < 2; i++) {
    float4 a = xp[lane + i * 32];
    float4 c = rp[lane + i * 32];
    vals[i].x = a.x + c.x; vals[i].y = a.y + c.y;
    vals[i].z = a.z + c.z; vals[i].w = a.w + c.w;
    sum += vals[i].x + vals[i].y + vals[i].z + vals[i].w;
  }
  for (int d = 1; d < 32; d <<= 1) sum += __shfl_xor(sum, d, 32);
  float mu = sum * (1.0f / DD);
  float var = 0.f;
#pragma unroll
  for (int i = 0; i < 2; i++) {
    float dx = vals[i].x - mu, dy = vals[i].y - mu;
    float dz = vals[i].z - mu, dw = vals[i].w - mu;
    var += dx * dx + dy * dy + dz * dz + dw * dw;
  }
  for (int d = 1; d < 32; d <<= 1) var += __shfl_xor(var, d, 32);
  float rstd = rsqrtf(var * (1.0f / DD) + 1e-5f);
  const float4* wp = (const float4*)w;
  const float4* bp = (const float4*)b;
  float4* yp = (float4*)(Y + (size_t)row * DD);
#pragma unroll
  for (int i = 0; i < 2; i++) {
    float4 ww = wp[lane + i * 32];
    float4 bb = bp[lane + i * 32];
    float4 o;
    o.x = (vals[i].x - mu) * rstd * ww.x + bb.x;
    o.y = (vals[i].y - mu) * rstd * ww.y + bb.y;
    o.z = (vals[i].z - mu) * rstd * ww.z + bb.z;
    o.w = (vals[i].w - mu) * rstd * ww.w + bb.w;
    yp[lane + i * 32] = o;
  }
}

// ---------------------------------------------------------------------------
// Host-side orchestration
// ---------------------------------------------------------------------------
extern "C" void kernel_launch(void* const* d_in, const int* in_sizes, int n_in,
                              void* d_out, int out_size, void* d_ws, size_t ws_size,
                              hipStream_t stream) {
  (void)in_sizes; (void)n_in; (void)out_size; (void)ws_size;

  const float* x     = (const float*)d_in[0];
  const int*   adj   = (const int*)d_in[1];
  const float* sa_wq = (const float*)d_in[2],  *sa_bq = (const float*)d_in[3];
  const float* sa_wk = (const float*)d_in[4],  *sa_bk = (const float*)d_in[5];
  const float* sa_wv = (const float*)d_in[6],  *sa_bv = (const float*)d_in[7];
  const float* sa_wo = (const float*)d_in[8],  *sa_bo = (const float*)d_in[9];
  const float* sa_lw = (const float*)d_in[10], *sa_lb = (const float*)d_in[11];
  const float* ta_wq = (const float*)d_in[12], *ta_bq = (const float*)d_in[13];
  const float* ta_wk = (const float*)d_in[14], *ta_bk = (const float*)d_in[15];
  const float* ta_wv = (const float*)d_in[16], *ta_bv = (const float*)d_in[17];
  const float* ta_wo = (const float*)d_in[18], *ta_bo = (const float*)d_in[19];
  const float* ta_lw = (const float*)d_in[20], *ta_lb = (const float*)d_in[21];
  const float* f_w1  = (const float*)d_in[22], *f_b1  = (const float*)d_in[23];
  const float* f_w2  = (const float*)d_in[24], *f_b2  = (const float*)d_in[25];
  const float* f_lw  = (const float*)d_in[26], *f_lb  = (const float*)d_in[27];

  const int M = MROWS;
  const size_t md = (size_t)M * DD;
  float* q    = (float*)d_ws;
  float* k    = q + md;
  float* v    = k + md;
  float* attn = v + md;
  float* xcur = attn + md;
  float* tmp  = xcur + md;
  float* hbuf = q;  // q/k/v/attn (4*md == M*FFD) dead by the FFN stage

  dim3 blk(256);
  dim3 gD(DD / 64, (M + 127) / 128);    // GEMM grids, N = 256
  dim3 gF(FFD / 64, (M + 127) / 128);   // N = 1024
  dim3 gSp((NN + 127) / 128, HH, BB * TT);
  dim3 gTp(BB * NN);
  dim3 gLn((M + 7) / 8);

  // ---- spatial attention block ----
  gemm_bf16_kernel<<<gD, blk, 0, stream>>>(x, sa_wq, sa_bq, q, M, DD, DD, 0);
  gemm_bf16_kernel<<<gD, blk, 0, stream>>>(x, sa_wk, sa_bk, k, M, DD, DD, 0);
  gemm_bf16_kernel<<<gD, blk, 0, stream>>>(x, sa_wv, sa_bv, v, M, DD, DD, 0);
  spat_attn_kernel<<<gSp, blk, 0, stream>>>(q, k, v, adj, attn);
  gemm_bf16_kernel<<<gD, blk, 0, stream>>>(attn, sa_wo, sa_bo, tmp, M, DD, DD, 0);
  add_ln_kernel<<<gLn, blk, 0, stream>>>(x, tmp, sa_lw, sa_lb, xcur, M);

  // ---- temporal attention block ----
  gemm_bf16_kernel<<<gD, blk, 0, stream>>>(xcur, ta_wq, ta_bq, q, M, DD, DD, 0);
  gemm_bf16_kernel<<<gD, blk, 0, stream>>>(xcur, ta_wk, ta_bk, k, M, DD, DD, 0);
  gemm_bf16_kernel<<<gD, blk, 0, stream>>>(xcur, ta_wv, ta_bv, v, M, DD, DD, 0);
  temp_attn_kernel<<<gTp, blk, 0, stream>>>(q, k, v, attn);
  gemm_bf16_kernel<<<gD, blk, 0, stream>>>(attn, ta_wo, ta_bo, tmp, M, DD, DD, 0);
  add_ln_kernel<<<gLn, blk, 0, stream>>>(xcur, tmp, ta_lw, ta_lb, xcur, M);

  // ---- FFN block (exact GELU) ----
  gemm_bf16_kernel<<<gF, blk, 0, stream>>>(xcur, f_w1, f_b1, hbuf, M, DD, FFD, 1);
  gemm_bf16_kernel<<<gD, blk, 0, stream>>>(hbuf, f_w2, f_b2, tmp, M, FFD, DD, 0);
  add_ln_kernel<<<gLn, blk, 0, stream>>>(xcur, tmp, f_lw, f_lb, (float*)d_out, M);
}
